// GraphSAGE_59133109732148
// MI455X (gfx1250) — matrix operand order repaired and checked by reading, not compiled, not run on hardware
//
#include <hip/hip_runtime.h>
#include <hip/hip_bf16.h>

// ---------- types ----------
typedef __attribute__((ext_vector_type(16))) __bf16 v16bf;
typedef __attribute__((ext_vector_type(8)))  __bf16 v8bf;
typedef __attribute__((ext_vector_type(8)))  float  v8f;

#define NNODES 100000
#define KNBR   10
#define DH     128          // D == H == O == 128
#define EPSLN  1e-5f

// LDS row strides padded to avoid 64-bank wraparound (256B stride -> 16-way conflict)
#define NB_STRIDE  136      // bf16 elems per row (128 + 8)
#define X_STRIDE   136
#define OUT_STRIDE 132      // f32 elems per row (128 + 4)

__device__ __forceinline__ v16bf cat8(v8bf lo, v8bf hi) {
    return __builtin_shufflevector(lo, hi, 0,1,2,3,4,5,6,7,8,9,10,11,12,13,14,15);
}

// ---------- prep kernels ----------
__global__ void cvt_f32_to_bf16(const float* __restrict__ in, __bf16* __restrict__ out, int n) {
    int i = blockIdx.x * 256 + threadIdx.x;
    if (i < n) out[i] = (__bf16)in[i];
}

// in: [indim][outdim] f32  ->  out: [outdim][indim] bf16
__global__ void transpose_to_bf16(const float* __restrict__ in, __bf16* __restrict__ out,
                                  int indim, int outdim) {
    int i = blockIdx.x * 256 + threadIdx.x;
    if (i < indim * outdim) {
        int o  = i % outdim;
        int ii = i / outdim;
        out[(size_t)o * indim + ii] = (__bf16)in[i];
    }
}

// ---------- one GraphSAGE layer: 16 nodes per block, 8 waves ----------
__global__ void __launch_bounds__(256)
sage_layer(const __bf16* __restrict__ xin,       // [N,128] bf16
           const int*    __restrict__ nbrs,      // [N,K]
           const __bf16* __restrict__ Pt,        // [128][128] bf16 (transposed: [out][in])
           const float*  __restrict__ pb,        // [128]
           const __bf16* __restrict__ Wt,        // [128][256] bf16 (transposed)
           const float*  __restrict__ wb,        // [128]
           const float*  __restrict__ lng,       // [128] or unused
           const float*  __restrict__ lnb,       // [128] or unused
           __bf16*       __restrict__ xout_bf,   // [N,128] bf16 (layers 0,1)
           float*        __restrict__ out_f32,   // [N,128] f32  (layer 2)
           int do_ln)
{
    __shared__ __bf16 nbLDS[KNBR * 16 * NB_STRIDE];   // gathered neighbors, k-major rows
    __shared__ __bf16 xLDS [16 * X_STRIDE];           // self features
    __shared__ __bf16 aggLDS[16 * X_STRIDE];          // max-pooled aggregate
    __shared__ float  outLDS[16 * OUT_STRIDE];        // pre-LN linear output
    __shared__ float  rowMean[16], rowInv[16];
    __shared__ int    sIdx[KNBR * 16];

    const int tid  = threadIdx.x;
    const int lane = tid & 31;
    const int wv   = tid >> 5;               // 0..7 : N-tile (16 output cols)
    const int mrow = lane & 15;              // row within 16-row tile
    const int nb0  = blockIdx.x * 16;        // first node of this block

    // ---- neighbor indices: row = k*16 + node (k-major!) ----
    if (tid < KNBR * 16) {
        int node = tid & 15, k = tid >> 4;
        sIdx[tid] = nbrs[(size_t)(nb0 + node) * KNBR + k];
    }
    __syncthreads();

    // ---- gather 160 neighbor rows (16B chunks) + 16 self rows into LDS ----
    #pragma unroll
    for (int j = 0; j < 10; ++j) {
        int flat = tid + j * 256;                    // 0..2559 : 160 rows * 16 chunks
        int row = flat >> 4, ch = flat & 15;
        *(v8bf*)(nbLDS + row * NB_STRIDE + ch * 8) =
            *(const v8bf*)(xin + (size_t)sIdx[row] * DH + ch * 8);
    }
    {
        int row = tid >> 4, ch = tid & 15;
        *(v8bf*)(xLDS + row * X_STRIDE + ch * 8) =
            *(const v8bf*)(xin + (size_t)(nb0 + row) * DH + ch * 8);
    }
    __syncthreads();

    const int colN   = wv * 16 + mrow;           // this lane's output column
    const int khalfA = (lane < 16) ? 0 : 8;      // A-frag K sub-offset
    const int khalfB = (lane < 16) ? 0 : 8;      // B-frag K offset

    // ---- pooling GEMM: agg = max_k relu(nb_k @ P + pb) ----
    // Hold the 4 B-fragments of Pt (this wave's 16 columns, K=128) in registers.
    v16bf bP[4];
    #pragma unroll
    for (int s = 0; s < 4; ++s) {
        const __bf16* p = Pt + (size_t)colN * DH + s * 32 + khalfB;
        bP[s] = cat8(*(const v8bf*)p, *(const v8bf*)(p + 16));
    }
    const float pbias = pb[colN];

    v8f agg = {};                                 // init 0 == relu absorbed into max
    #pragma unroll
    for (int t = 0; t < KNBR; ++t) {              // M-tile t == neighbor slot k=t
        v8f acc = {};
        #pragma unroll
        for (int s = 0; s < 4; ++s) {
            const __bf16* ap = nbLDS + (t * 16 + mrow) * NB_STRIDE + s * 32 + khalfA;
            v16bf a = cat8(*(const v8bf*)ap, *(const v8bf*)(ap + 16));
            acc = __builtin_amdgcn_wmma_f32_16x16x32_bf16(
                      false, a, false, bP[s], (short)0, acc, false, false);
        }
        #pragma unroll
        for (int r = 0; r < 8; ++r)
            agg[r] = fmaxf(agg[r], acc[r] + pbias);
    }

    // ---- agg tile -> LDS as bf16 (C layout: row = r + 8*(lane>=16), col = colN) ----
    #pragma unroll
    for (int r = 0; r < 8; ++r) {
        int row = r + ((lane < 16) ? 0 : 8);
        aggLDS[row * X_STRIDE + colN] = (__bf16)agg[r];
    }
    __syncthreads();

    // ---- main GEMM: out = [x | agg] @ W + wb   (M=16, K=256, this wave: 16 cols) ----
    v8f acc = {};
    #pragma unroll
    for (int s = 0; s < 8; ++s) {
        const __bf16* asrc = (s < 4) ? (xLDS   + mrow * X_STRIDE + s * 32)
                                     : (aggLDS + mrow * X_STRIDE + (s - 4) * 32);
        const __bf16* ap = asrc + khalfA;
        v16bf a = cat8(*(const v8bf*)ap, *(const v8bf*)(ap + 16));
        const __bf16* wp = Wt + (size_t)colN * 256 + s * 32 + khalfB;
        v16bf b = cat8(*(const v8bf*)wp, *(const v8bf*)(wp + 16));
        acc = __builtin_amdgcn_wmma_f32_16x16x32_bf16(
                  false, a, false, b, (short)0, acc, false, false);
    }
    const float wbias = wb[colN];
    #pragma unroll
    for (int r = 0; r < 8; ++r) acc[r] += wbias;

    if (do_ln) {
        // stage fp32 tile, compute per-row LN stats, apply LN+relu+residual, emit bf16
        #pragma unroll
        for (int r = 0; r < 8; ++r) {
            int row = r + ((lane < 16) ? 0 : 8);
            outLDS[row * OUT_STRIDE + colN] = acc[r];
        }
        __syncthreads();
        if (tid < 16) {
            float s = 0.f, sq = 0.f;
            #pragma unroll 4
            for (int c = 0; c < DH; ++c) {
                float v = outLDS[tid * OUT_STRIDE + c];
                s += v; sq += v * v;
            }
            float m   = s * (1.0f / DH);
            float var = sq * (1.0f / DH) - m * m;
            rowMean[tid] = m;
            rowInv[tid]  = rsqrtf(var + EPSLN);
        }
        __syncthreads();
        #pragma unroll
        for (int j = 0; j < 8; ++j) {
            int e   = tid + j * 256;         // 0..2047
            int row = e >> 7, col = e & 127;
            float v  = outLDS[row * OUT_STRIDE + col];
            float ln = (v - rowMean[row]) * rowInv[row] * lng[col] + lnb[col];
            float y  = fmaxf(ln, 0.f) + v;   // relu(LN(x)) + residual
            xout_bf[(size_t)(nb0 + row) * DH + col] = (__bf16)y;
        }
    } else {
        #pragma unroll
        for (int r = 0; r < 8; ++r) {
            int row = r + ((lane < 16) ? 0 : 8);
            out_f32[(size_t)(nb0 + row) * DH + colN] = acc[r];
        }
    }
}

// ---------- launcher ----------
extern "C" void kernel_launch(void* const* d_in, const int* in_sizes, int n_in,
                              void* d_out, int out_size, void* d_ws, size_t ws_size,
                              hipStream_t stream) {
    const float* features = (const float*)d_in[0];
    const int*   neighbors= (const int*)  d_in[1];
    const float* W0 = (const float*)d_in[2];  const float* b0  = (const float*)d_in[3];
    const float* W1 = (const float*)d_in[4];  const float* b1  = (const float*)d_in[5];
    const float* W2 = (const float*)d_in[6];  const float* b2  = (const float*)d_in[7];
    const float* P0 = (const float*)d_in[8];  const float* pb0 = (const float*)d_in[9];
    const float* P1 = (const float*)d_in[10]; const float* pb1 = (const float*)d_in[11];
    const float* P2 = (const float*)d_in[12]; const float* pb2 = (const float*)d_in[13];
    const float* ln0g = (const float*)d_in[14]; const float* ln0b = (const float*)d_in[15];
    const float* ln1g = (const float*)d_in[16]; const float* ln1b = (const float*)d_in[17];
    float* out = (float*)d_out;

    // workspace layout: two bf16 activation buffers + transposed bf16 weights
    const size_t XELEMS = (size_t)NNODES * DH;        // 12.8M
    char* ws = (char*)d_ws;
    __bf16* xA  = (__bf16*)ws;
    __bf16* xB  = (__bf16*)(ws + XELEMS * sizeof(__bf16));
    __bf16* Wt0 = (__bf16*)(ws + 2 * XELEMS * sizeof(__bf16));
    __bf16* Wt1 = Wt0 + (size_t)256 * 128;
    __bf16* Wt2 = Wt1 + (size_t)256 * 128;
    __bf16* Pt0 = Wt2 + (size_t)256 * 128;
    __bf16* Pt1 = Pt0 + (size_t)128 * 128;
    __bf16* Pt2 = Pt1 + (size_t)128 * 128;

    // prep: features -> bf16; weights -> transposed bf16 [out][in]
    {
        int n = (int)XELEMS;
        cvt_f32_to_bf16<<<(n + 255) / 256, 256, 0, stream>>>(features, xA, n);
        int nw = 256 * 128;  // W matrices
        transpose_to_bf16<<<(nw + 255) / 256, 256, 0, stream>>>(W0, Wt0, 256, 128);
        transpose_to_bf16<<<(nw + 255) / 256, 256, 0, stream>>>(W1, Wt1, 256, 128);
        transpose_to_bf16<<<(nw + 255) / 256, 256, 0, stream>>>(W2, Wt2, 256, 128);
        int np = 128 * 128;  // P matrices
        transpose_to_bf16<<<(np + 255) / 256, 256, 0, stream>>>(P0, Pt0, 128, 128);
        transpose_to_bf16<<<(np + 255) / 256, 256, 0, stream>>>(P1, Pt1, 128, 128);
        transpose_to_bf16<<<(np + 255) / 256, 256, 0, stream>>>(P2, Pt2, 128, 128);
    }

    dim3 grid(NNODES / 16), block(256);
    // layer 0: xA -> xB (LN+relu+residual)
    sage_layer<<<grid, block, 0, stream>>>(xA, neighbors, Pt0, pb0, Wt0, b0,
                                           ln0g, ln0b, xB, nullptr, 1);
    // layer 1: xB -> xA
    sage_layer<<<grid, block, 0, stream>>>(xB, neighbors, Pt1, pb1, Wt1, b1,
                                           ln1g, ln1b, xA, nullptr, 1);
    // layer 2: xA -> d_out (f32, no LN)
    sage_layer<<<grid, block, 0, stream>>>(xA, neighbors, Pt2, pb2, Wt2, b2,
                                           nullptr, nullptr, nullptr, out, 0);
}
